// ColbertModel_1288490189594
// MI455X (gfx1250) — matrix operand order repaired and checked by reading, not compiled, not run on hardware
//
#include <hip/hip_runtime.h>
#include <math.h>

// ---- types matching gfx1250 WMMA builtin signatures (probe-confirmed) ----
typedef __attribute__((ext_vector_type(16))) __bf16 v16bf;
typedef __attribute__((ext_vector_type(8)))  __bf16 v8bf;
typedef __attribute__((ext_vector_type(8)))  float  v8f;

#define DIM 128
#define HID 768
#define BQ 64
#define LQ 32
#define BP 64
#define LP 512
#define NQROWS (BQ * LQ)            // 2048
#define NPROWS (BP * LP)            // 32768
#define NROWS  (NQROWS + NPROWS)    // 34816
#define NTILES (NROWS / 16)         // 2176

// load 8 consecutive fp32 and convert to bf16 (compiler vectorizes to b128 loads)
__device__ __forceinline__ v8bf load_cvt8(const float* __restrict__ p) {
  v8bf r;
#pragma unroll
  for (int j = 0; j < 8; ++j) r[j] = (__bf16)p[j];
  return r;
}

__device__ __forceinline__ v16bf cat16(v8bf lo, v8bf hi) {
  return __builtin_shufflevector(lo, hi, 0, 1, 2, 3, 4, 5, 6, 7,
                                 8, 9, 10, 11, 12, 13, 14, 15);
}

// ------------------------------------------------------------------
// Kernel 0: W fp32 -> bf16
// ------------------------------------------------------------------
__global__ void cvt_w_kernel(const float* __restrict__ W, __bf16* __restrict__ Wb) {
  int i = blockIdx.x * blockDim.x + threadIdx.x;
  if (i < DIM * HID) Wb[i] = (__bf16)W[i];
}

// ------------------------------------------------------------------
// Kernel 1: projection (rows x 768) @ W^T (768 x 128) + L2 norm -> bf16
// One wave per 16-row tile; 4 waves per block.
// A-fragment layout (16-bit A 16x32):
//   lane<16: row=lane, elems 0..15 = K kb..kb+15 contiguous (kb = ks*32)
//   lane>=16: same rows, K shifted by +16.
// B-fragment layout (16-bit B 32x16):
//   lane<16: col=lane, K 0..15 contiguous; lane>=16: col=lane-16, K 16..31.
// ------------------------------------------------------------------
__global__ void __launch_bounds__(128)
proj_norm_kernel(const float* __restrict__ qh, const float* __restrict__ ph,
                 const __bf16* __restrict__ Wb,
                 __bf16* __restrict__ Qb, __bf16* __restrict__ Db) {
  __shared__ float tile[4][16][DIM + 4];
  __shared__ float rnormS[4][16];

  const int wave = threadIdx.x >> 5;
  const int lane = threadIdx.x & 31;
  const int half = lane >> 4;
  const int l15  = lane & 15;
  const int tileId = blockIdx.x * 4 + wave;
  const int r0 = tileId * 16;

  const float* base = (r0 < NQROWS) ? (qh + (size_t)r0 * HID)
                                    : (ph + (size_t)(r0 - NQROWS) * HID);
  const float* rp = base + (size_t)l15 * HID;

  v8f acc[8] = {};

  for (int ks = 0; ks < HID / 32; ++ks) {
    const int kb = ks * 32 + half * 16;
    v16bf a = cat16(load_cvt8(rp + kb), load_cvt8(rp + kb + 8));
    const int kb2 = ks * 32 + half * 16;
#pragma unroll
    for (int c = 0; c < 8; ++c) {
      const __bf16* wp = Wb + (size_t)(c * 16 + l15) * HID + kb2;  // 32B aligned
      v16bf b = *(const v16bf*)wp;
      acc[c] = __builtin_amdgcn_wmma_f32_16x16x32_bf16(
          false, a, false, b, (short)0, acc[c], false, false);
    }
  }

  // C/D layout: vgpr i -> row (i + half*8), col = c*16 + l15
#pragma unroll
  for (int c = 0; c < 8; ++c)
#pragma unroll
    for (int i = 0; i < 8; ++i)
      tile[wave][i + half * 8][c * 16 + l15] = acc[c][i];
  __syncthreads();

  if (lane < 16) {
    float s = 0.0f;
#pragma unroll 8
    for (int c = 0; c < DIM; ++c) { float v = tile[wave][l15][c]; s += v * v; }
    rnormS[wave][l15] = 1.0f / fmaxf(sqrtf(s), 1e-12f);
  }
  __syncthreads();

  for (int idx = lane; idx < 16 * DIM; idx += 32) {
    const int r  = idx >> 7;
    const int cc = idx & (DIM - 1);
    const float v = tile[wave][r][cc] * rnormS[wave][r];
    const int grow = r0 + r;
    if (grow < NQROWS) Qb[(size_t)grow * DIM + cc] = (__bf16)v;
    else               Db[(size_t)(grow - NQROWS) * DIM + cc] = (__bf16)v;
  }
}

// ------------------------------------------------------------------
// Kernel 2: MaxSim. One wave per (q,p) pair; 4 waves/block.
// S = Q_q (32x128) @ D_p^T (128x512); running max over doc columns kept in
// the C/D accumulator layout; cross-lane max (xor 1,2,4,8 within each
// 16-lane half) -> row maxes; sum over 32 query rows -> score[q][p].
// ------------------------------------------------------------------
__global__ void __launch_bounds__(128)
maxsim_kernel(const __bf16* __restrict__ Qb, const __bf16* __restrict__ Db,
              float* __restrict__ out) {
  const int wave = threadIdx.x >> 5;
  const int lane = threadIdx.x & 31;
  const int half = lane >> 4;
  const int l15  = lane & 15;
  const int q = blockIdx.x;
  const int p = blockIdx.y * 4 + wave;

  const __bf16* Qbase = Qb + (size_t)q * LQ * DIM;
  const __bf16* Dbase = Db + (size_t)p * LP * DIM;

  // resident Q fragments: 2 row-tiles x 4 K-steps
  v16bf afrag[2][4];
#pragma unroll
  for (int t = 0; t < 2; ++t)
#pragma unroll
    for (int ks = 0; ks < 4; ++ks) {
      const __bf16* ap = Qbase + (size_t)(t * 16 + l15) * DIM + ks * 32 + half * 16;
      afrag[t][ks] = cat16(*(const v8bf*)ap, *(const v8bf*)(ap + 8));
    }

  v8f vmax0, vmax1;
#pragma unroll
  for (int i = 0; i < 8; ++i) { vmax0[i] = -1e30f; vmax1[i] = -1e30f; }

  for (int ct = 0; ct < LP / 16; ++ct) {
    v8f a0 = {}, a1 = {};
    const __bf16* dcol = Dbase + (size_t)(ct * 16 + l15) * DIM + half * 16;
#pragma unroll
    for (int ks = 0; ks < 4; ++ks) {
      v16bf b = *(const v16bf*)(dcol + ks * 32);  // 32B aligned
      a0 = __builtin_amdgcn_wmma_f32_16x16x32_bf16(
          false, afrag[0][ks], false, b, (short)0, a0, false, false);
      a1 = __builtin_amdgcn_wmma_f32_16x16x32_bf16(
          false, afrag[1][ks], false, b, (short)0, a1, false, false);
    }
#pragma unroll
    for (int i = 0; i < 8; ++i) {
      vmax0[i] = fmaxf(vmax0[i], a0[i]);
      vmax1[i] = fmaxf(vmax1[i], a1[i]);
    }
  }

  // reduce: max over the 16 lanes of each half (cols), then sum rows
  float sum = 0.0f;
#pragma unroll
  for (int i = 0; i < 8; ++i) {
    float m0 = vmax0[i], m1 = vmax1[i];
#pragma unroll
    for (int mask = 1; mask <= 8; mask <<= 1) {
      m0 = fmaxf(m0, __shfl_xor(m0, mask, 32));
      m1 = fmaxf(m1, __shfl_xor(m1, mask, 32));
    }
    sum += m0 + m1;  // rows {16t + i + half*8}
  }
  sum += __shfl_xor(sum, 16, 32);  // combine both lane halves (all 32 rows)

  if (lane == 0) out[q * BP + p] = sum;
}

// ------------------------------------------------------------------
extern "C" void kernel_launch(void* const* d_in, const int* in_sizes, int n_in,
                              void* d_out, int out_size, void* d_ws, size_t ws_size,
                              hipStream_t stream) {
  const float* qh = (const float*)d_in[0];  // [64,32,768]
  const float* ph = (const float*)d_in[1];  // [64,512,768]
  const float* W  = (const float*)d_in[2];  // [128,768]
  float* out = (float*)d_out;               // [64,64]

  char* ws = (char*)d_ws;
  __bf16* Wb = (__bf16*)ws;                                     // 196,608 B
  __bf16* Qb = (__bf16*)(ws + 196608);                          // 524,288 B
  __bf16* Db = (__bf16*)(ws + 196608 + 524288);                 // 8,388,608 B

  cvt_w_kernel<<<(DIM * HID + 255) / 256, 256, 0, stream>>>(W, Wb);
  proj_norm_kernel<<<NTILES / 4, 128, 0, stream>>>(qh, ph, Wb, Qb, Db);
  maxsim_kernel<<<dim3(BQ, BP / 4), 128, 0, stream>>>(Qb, Db, out);
}
